// MultiHeadAttention_29798483100289
// MI455X (gfx1250) — matrix operand order repaired and checked by reading, not compiled, not run on hardware
//
#include <hip/hip_runtime.h>
#include <hip/hip_bf16.h>

// Problem constants (match reference)
#define BB 4
#define SS 2048
#define EE 1024
#define HH 16
#define DD 64

typedef __bf16 bf16_t;
typedef __attribute__((ext_vector_type(16))) __bf16 v16bf;
typedef __attribute__((ext_vector_type(8)))  __bf16 v8bf;
typedef __attribute__((ext_vector_type(8)))  float  v8f;

static __device__ __forceinline__ v8f wmma_bf16(v16bf a, v16bf b, v8f c) {
  // (neg_a, A, neg_b, B, c_mod, C, reuse_a, reuse_b)
  return __builtin_amdgcn_wmma_f32_16x16x32_bf16(false, a, false, b, (short)0, c, false, false);
}

static __device__ __forceinline__ v8f vzero8() {
  v8f z = {0.f, 0.f, 0.f, 0.f, 0.f, 0.f, 0.f, 0.f};
  return z;
}

// A-fragment (16x32 bf16): lane holds row m=lane&15;
// k indices = 2*base8+0..15, base8 = 8*(lane>>4)
// => one contiguous 32B run at rowp[k0+2*base8].
static __device__ __forceinline__ v16bf load_a_bf16(const bf16_t* __restrict__ rowp,
                                                    int k0, int base8) {
  v16bf a;
  const bf16_t* p = rowp + k0 + 2 * base8;
#pragma unroll
  for (int j = 0; j < 8; ++j) a[j] = p[j];
#pragma unroll
  for (int j = 0; j < 8; ++j) a[j + 8] = p[j + 8];
  return a;
}

// B-fragment: 16 contiguous bf16 per lane, read as 2 x 16B.
static __device__ __forceinline__ v16bf load_b_2x16(const bf16_t* __restrict__ p) {
  const v8bf* q = (const v8bf*)p;
  v8bf x = q[0], y = q[1];
  v16bf b;
#pragma unroll
  for (int j = 0; j < 8; ++j) { b[j] = x[j]; b[j + 8] = y[j]; }
  return b;
}

// CDNA5 async copy: 16B global -> LDS per lane, tracked by ASYNCcnt.
static __device__ __forceinline__ void async_copy_b128(unsigned lds_off,
                                                       const bf16_t* g) {
  asm volatile("global_load_async_to_lds_b128 %0, %1, off"
               :: "v"(lds_off), "v"(g) : "memory");
}
static __device__ __forceinline__ void wait_async0() {
  asm volatile("s_wait_asynccnt 0" ::: "memory");
}

// ---------------------------------------------------------------------------
// fp32 -> bf16 elementwise convert (weights + activations, done once)
// ---------------------------------------------------------------------------
__global__ void __launch_bounds__(256)
cvt_f32_bf16(const float* __restrict__ in, bf16_t* __restrict__ out, int n) {
  int i = blockIdx.x * 256 + threadIdx.x;
  if (i < n) out[i] = (bf16_t)in[i];
}

// ---------------------------------------------------------------------------
// GEMM: Out[m,n] = sum_k A[m,k] * W[n,k]   (A @ W^T, Linear semantics)
//   A: [M,K] bf16;  W: [N,K] bf16
// Block = 8 waves sharing one 64-wide N strip. The 64x32 W tile per k-step is
// staged cooperatively into LDS with GLOBAL_LOAD_ASYNC_TO_LDS_B128 (double
// buffered; one 16B async op per thread per step). Each wave computes a 32x64
// tile: 2 A-frags x 4 LDS B-frags -> 8 WMMA per k-step, issued as one block
// after all fragment loads so VMEM/DS latency is covered by one wait.
// OMODE: 0 = bf16 [B,H,S,D] (Q,K) | 1 = bf16 [B,H,D,S] (V^T) | 2 = f32 [M,N]
// ---------------------------------------------------------------------------
template <int OMODE>
__global__ void __launch_bounds__(256)
gemm_wmma(const bf16_t* __restrict__ A, const bf16_t* __restrict__ W,
          void* __restrict__ Outv, int M, int N, int K) {
  // 80B row stride (64B data + 16B pad) -> conflict-free 32B fragment reads
  __shared__ bf16_t wt[2][64][40];

  const int tid  = threadIdx.x;
  const int lane = tid & 31;
  const int wid  = tid >> 5;
  const int bx   = blockIdx.x;
  const int nstr = N >> 6;
  const int nb   = bx & (nstr - 1);          // N strip (shared by block)
  const int mtb  = bx / nstr;                // block row-group
  const int mrow0 = (mtb * 8 + wid) * 32;    // wave's 32-row base

  const int lo    = lane & 15;
  const int base8 = (lane & 16) ? 8 : 0;     // A k-offset / C row-offset
  const int k16   = (lane & 16) ? 16 : 0;    // B k-offset

  // cooperative staging assignment: thread -> (row, 16B chunk)
  const int srow   = tid >> 2;               // 0..63
  const int schunk = tid & 3;                // 0..3
  const bf16_t* gW = W + (size_t)(nb * 64 + srow) * K + schunk * 8;
  const unsigned lds0 =
      (unsigned)(unsigned long long)&wt[0][srow][schunk * 8];
  const unsigned lds1 =
      (unsigned)(unsigned long long)&wt[1][srow][schunk * 8];

  const bf16_t* arow0 = A + (size_t)(mrow0 + lo) * K;
  const bf16_t* arow1 = A + (size_t)(mrow0 + 16 + lo) * K;

  v8f acc[2][4];
#pragma unroll
  for (int h = 0; h < 2; ++h)
#pragma unroll
    for (int t = 0; t < 4; ++t) acc[h][t] = vzero8();

  // prologue: stage k0 = 0 into buffer 0
  async_copy_b128(lds0, gW);
  wait_async0();
  __syncthreads();

  for (int k0 = 0; k0 < K; k0 += 32) {
    const int cur = (k0 >> 5) & 1;
    if (k0 + 32 < K)  // stage next tile into the other buffer (no wait yet)
      async_copy_b128(cur ? lds0 : lds1, gW + (k0 + 32));

    // fetch all fragments first, then issue the 8 WMMAs back-to-back
    const v16bf a0 = load_a_bf16(arow0, k0, base8);
    const v16bf a1 = load_a_bf16(arow1, k0, base8);
    v16bf bfrag[4];
#pragma unroll
    for (int t = 0; t < 4; ++t)
      bfrag[t] = load_b_2x16(&wt[cur][t * 16 + lo][k16]);

#pragma unroll
    for (int t = 0; t < 4; ++t) {
      acc[0][t] = wmma_bf16(a0, bfrag[t], acc[0][t]);
      acc[1][t] = wmma_bf16(a1, bfrag[t], acc[1][t]);
    }
    wait_async0();     // our slice of next tile is in LDS
    __syncthreads();   // everyone's slice is in LDS; safe to reuse buffers
  }

  // C/D layout: lane holds col n = lo; VGPR r holds row r + base8.
#pragma unroll
  for (int h = 0; h < 2; ++h) {
#pragma unroll
    for (int t = 0; t < 4; ++t) {
#pragma unroll
      for (int r = 0; r < 8; ++r) {
        const int m   = mrow0 + h * 16 + r + base8;
        const int col = nb * 64 + t * 16 + lo;
        const float val = acc[h][t][r];
        if constexpr (OMODE == 0) {
          const int b = m >> 11, s = m & (SS - 1);
          const int hh = col >> 6, d = col & 63;
          ((bf16_t*)Outv)[(((size_t)(b * HH + hh)) * SS + s) * DD + d] = (bf16_t)val;
        } else if constexpr (OMODE == 1) {
          const int b = m >> 11, s = m & (SS - 1);
          const int hh = col >> 6, d = col & 63;
          ((bf16_t*)Outv)[(((size_t)(b * HH + hh)) * DD + d) * SS + s] = (bf16_t)val;
        } else {
          ((float*)Outv)[(size_t)m * N + col] = val;
        }
      }
    }
  }
}

// ---------------------------------------------------------------------------
// Flash attention (causal): one wave per 16-query tile of one (b,h).
//   Qb, Kb: bf16 [B,H,S,D]; Vt: bf16 [B,H,D,S]; Ob: bf16 [B,S,E]
// kv blocks of 32 -> 4 wmma (scores) + 4 wmma (P.V) per block. All four K
// fragments are loaded before the score WMMAs (single clause + single wait);
// V fragments are loaded before the P re-layout so global latency overlaps
// the LDS traffic. Next kv block is prefetched via global_prefetch_b8.
// ---------------------------------------------------------------------------
__global__ void __launch_bounds__(32)
attn_fa(const bf16_t* __restrict__ Qb, const bf16_t* __restrict__ Kb,
        const bf16_t* __restrict__ Vt, bf16_t* __restrict__ Ob) {
  __shared__ bf16_t Pl[16][40];

  const int lane  = threadIdx.x & 31;
  const int lo    = lane & 15;
  const int base8 = (lane & 16) ? 8 : 0;
  const int k16   = (lane & 16) ? 16 : 0;

  const int g     = blockIdx.x;
  const int qt    = g & (SS / 16 - 1);
  const int bh    = g >> 7;                 // b*H + h
  const int qbase = qt * 16;

  const bf16_t* Qh = Qb + (size_t)bh * SS * DD;
  const bf16_t* Kh = Kb + (size_t)bh * SS * DD;
  const bf16_t* Vh = Vt + (size_t)bh * DD * SS;

  const bf16_t* qrow = Qh + (size_t)(qbase + lo) * DD;
  const v16bf aq0 = load_a_bf16(qrow, 0, base8);
  const v16bf aq1 = load_a_bf16(qrow, 32, base8);

  float mrow[8], lrow[8];
  v8f o[4];
#pragma unroll
  for (int r = 0; r < 8; ++r) { mrow[r] = -1e30f; lrow[r] = 0.f; }
#pragma unroll
  for (int t = 0; t < 4; ++t) o[t] = vzero8();

  const int jbmax = (qbase + 15) >> 5;
  for (int jb = 0; jb <= jbmax; ++jb) {
    const int kvb = jb << 5;

    // prefetch next kv block's K and V rows into cache
    if (jb < jbmax) {
      __builtin_prefetch(Kh + (size_t)(kvb + 32 + lo) * DD, 0, 1);
      __builtin_prefetch(Kh + (size_t)(kvb + 48 + lo) * DD, 0, 1);
#pragma unroll
      for (int dt = 0; dt < 4; ++dt)
        __builtin_prefetch(Vh + (size_t)(dt * 16 + lo) * SS + kvb + 32, 0, 1);
    }

    // ---- scores S = (Q K^T) for 16 q x 32 kv: load all frags, then MMA ---
    const bf16_t* kr0 = Kh + (size_t)(kvb + lo) * DD + k16;
    const bf16_t* kr1 = Kh + (size_t)(kvb + 16 + lo) * DD + k16;
    const v16bf b00 = load_b_2x16(kr0);
    const v16bf b01 = load_b_2x16(kr0 + 32);
    const v16bf b10 = load_b_2x16(kr1);
    const v16bf b11 = load_b_2x16(kr1 + 32);

    v8f s0 = vzero8(), s1 = vzero8();
    s0 = wmma_bf16(aq0, b00, s0);
    s0 = wmma_bf16(aq1, b01, s0);
    s1 = wmma_bf16(aq0, b10, s1);
    s1 = wmma_bf16(aq1, b11, s1);

    // ---- scale, causal mask, online softmax ------------------------------
#pragma unroll
    for (int r = 0; r < 8; ++r) {
      const int qi = qbase + r + base8;
      float v0 = s0[r] * 0.125f;                 // 1/sqrt(64)
      if (kvb + lo > qi)      v0 = -1e30f;
      float v1 = s1[r] * 0.125f;
      if (kvb + 16 + lo > qi) v1 = -1e30f;

      float mx = fmaxf(v0, v1);
#pragma unroll
      for (int off = 1; off < 16; off <<= 1)
        mx = fmaxf(mx, __shfl_xor(mx, off, 16));

      const float nm = fmaxf(mrow[r], mx);
      const float fs = __expf(mrow[r] - nm);
      const float p0 = __expf(v0 - nm);
      const float p1 = __expf(v1 - nm);
      float sum = p0 + p1;
#pragma unroll
      for (int off = 1; off < 16; off <<= 1)
        sum += __shfl_xor(sum, off, 16);

      lrow[r] = lrow[r] * fs + sum;
      mrow[r] = nm;
#pragma unroll
      for (int t = 0; t < 4; ++t) o[t][r] = o[t][r] * fs;

      Pl[r + base8][lo]      = (bf16_t)p0;
      Pl[r + base8][16 + lo] = (bf16_t)p1;
    }

    // ---- V fragments: issue global loads before the LDS round-trip -------
    v16bf vb[4];
#pragma unroll
    for (int dt = 0; dt < 4; ++dt)
      vb[dt] = load_b_2x16(Vh + (size_t)(dt * 16 + lo) * SS + kvb + k16);

    __syncthreads();

    // ---- re-layout P into A-fragment, accumulate O += P.V ----------------
    v16bf ap;
#pragma unroll
    for (int j = 0; j < 16; ++j) {
      const int kk = 2 * base8 + j;
      ap[j] = Pl[lo][kk];
    }
#pragma unroll
    for (int dt = 0; dt < 4; ++dt)
      o[dt] = wmma_bf16(ap, vb[dt], o[dt]);

    __syncthreads();
  }

  const int b = bh >> 4, h = bh & 15;
#pragma unroll
  for (int dt = 0; dt < 4; ++dt) {
#pragma unroll
    for (int r = 0; r < 8; ++r) {
      const int sidx = qbase + r + base8;
      const float val = o[dt][r] / lrow[r];
      Ob[((size_t)b * SS + sidx) * EE + h * DD + dt * 16 + lo] = (bf16_t)val;
    }
  }
}

// ---------------------------------------------------------------------------
extern "C" void kernel_launch(void* const* d_in, const int* in_sizes, int n_in,
                              void* d_out, int out_size, void* d_ws, size_t ws_size,
                              hipStream_t stream) {
  (void)in_sizes; (void)n_in; (void)out_size; (void)ws_size;

  const float* query = (const float*)d_in[0];
  const float* key   = (const float*)d_in[1];
  const float* value = (const float*)d_in[2];
  const float* Wq    = (const float*)d_in[3];
  const float* Wk    = (const float*)d_in[4];
  const float* Wv    = (const float*)d_in[5];
  const float* Wo    = (const float*)d_in[6];
  // d_in[7] = causal mask, applied analytically in-kernel.

  // Workspace (bf16): 4 weights (2MB) + 3 converted activations (16MB)
  // + Q,K,Vt,O (16MB each)  => ~120MB total
  char* w = (char*)d_ws;
  const size_t wsz = (size_t)EE * EE;       // 1,048,576 elems
  const size_t qsz = (size_t)BB * SS * EE;  // 8,388,608 elems
  bf16_t* Wqb = (bf16_t*)w; w += wsz * 2;
  bf16_t* Wkb = (bf16_t*)w; w += wsz * 2;
  bf16_t* Wvb = (bf16_t*)w; w += wsz * 2;
  bf16_t* Wob = (bf16_t*)w; w += wsz * 2;
  bf16_t* Xq  = (bf16_t*)w; w += qsz * 2;
  bf16_t* Xk  = (bf16_t*)w; w += qsz * 2;
  bf16_t* Xv  = (bf16_t*)w; w += qsz * 2;
  bf16_t* Qb  = (bf16_t*)w; w += qsz * 2;
  bf16_t* Kb  = (bf16_t*)w; w += qsz * 2;
  bf16_t* Vt  = (bf16_t*)w; w += qsz * 2;
  bf16_t* Ob  = (bf16_t*)w; w += qsz * 2;

  const int nw = (int)wsz, na = (int)qsz;
  cvt_f32_bf16<<<(nw + 255) / 256, 256, 0, stream>>>(Wq, Wqb, nw);
  cvt_f32_bf16<<<(nw + 255) / 256, 256, 0, stream>>>(Wk, Wkb, nw);
  cvt_f32_bf16<<<(nw + 255) / 256, 256, 0, stream>>>(Wv, Wvb, nw);
  cvt_f32_bf16<<<(nw + 255) / 256, 256, 0, stream>>>(Wo, Wob, nw);
  cvt_f32_bf16<<<(na + 255) / 256, 256, 0, stream>>>(query, Xq, na);
  cvt_f32_bf16<<<(na + 255) / 256, 256, 0, stream>>>(key,   Xk, na);
  cvt_f32_bf16<<<(na + 255) / 256, 256, 0, stream>>>(value, Xv, na);

  const int M = BB * SS, N = EE, K = EE;
  // blocks: (M/32/8 row-groups) x (N/64 strips) = 32 x 16 = 512
  const int blocks = (M / 256) * (N / 64);

  gemm_wmma<0><<<blocks, 256, 0, stream>>>(Xq, Wqb, Qb, M, N, K);
  gemm_wmma<0><<<blocks, 256, 0, stream>>>(Xk, Wkb, Kb, M, N, K);
  gemm_wmma<1><<<blocks, 256, 0, stream>>>(Xv, Wvb, Vt, M, N, K);

  attn_fa<<<BB * HH * (SS / 16), 32, 0, stream>>>(Qb, Kb, Vt, Ob);

  gemm_wmma<2><<<blocks, 256, 0, stream>>>(Ob, Wob, d_out, M, N, K);
}
